// GatedDynamicRoutingRNN_11622181503022
// MI455X (gfx1250) — matrix operand order repaired and checked by reading, not compiled, not run on hardware
//
#include <hip/hip_runtime.h>
#include <hip/hip_bf16.h>
#include <stdint.h>

// ---------------------------------------------------------------------------
// Gated RNN on MI455X (gfx1250, wave32), persistent-kernel version.
//   per step: pre  = y @ W_rec.T  + wu_t
//             gate = sigmoid(y @ W_gate.T + wug_t)
//             y'   = gate*y + (1-gate)*relu(pre) + NOISE_SCALE*n_t
//
// Design points (from MI455X specs):
//  * bf16 WMMA (v_wmma_f32_16x16x32_bf16) for both per-step GEMMs, fp32 state.
//  * ONE kernel for all 512 steps: device-scope atomic grid barrier between
//    steps (512 separate launches would dominate the ~0.1-0.5us/step of work).
//  * 320KB LDS/WGP: each block keeps its 256KB packed bf16 weight strip
//    resident in LDS for the whole run -> weights read from L2 exactly once.
//  * B-fragments stored as split 16B halves so ds_load_b128 lane stride is
//    16B (4 banks x 16 lanes = 64 banks, conflict-free phases).
//  * Inner K loop is explicitly double-buffered: tile kt+1's 6 LDS loads are
//    issued before tile kt's two WMMAs, so ds_load latency is hidden behind
//    matrix math (only ~1 wave/SIMD is resident in this persistent config,
//    so the wave must do its own latency hiding).
// ---------------------------------------------------------------------------

typedef __bf16 v16bf __attribute__((ext_vector_type(16)));
typedef __bf16 v8bf  __attribute__((ext_vector_type(8)));
typedef float  v8f   __attribute__((ext_vector_type(8)));

#define B_  128
#define T_  512
#define N_  1024
#define NI_ 7
#define NTILES (N_ / 16)   // 64 n-tiles
#define KTILES (N_ / 32)   // 32 k-tiles of 32
#define NOISE_SCALE 0.09486832980505138f   // 0.15 * sqrt(2*0.2)
#define GRID_X 16
#define GRID_Y 8
#define NBLOCKS (GRID_X * GRID_Y)          // 128 persistent blocks

// dynamic LDS layout (bytes)
#define SY_BYTES   (16 * (N_ + 8) * 2)     // 33024: y tile, bf16, padded rows
#define SU_OFF     SY_BYTES
#define SU_BYTES   (16 * 8 * 4)            // 512
#define SW_OFF     (SU_OFF + SU_BYTES)     // wave weight strips start
#define SW_PER_WAVE 65536                  // 2 matrices x 32KB
#define SMEM_BYTES (SW_OFF + 4 * SW_PER_WAVE)  // 295,680 < 320KB/WGP

// ---------------------------------------------------------------------------
// Pack fp32 weight matrix W[N][N] (row n, col k) into bf16 B-fragments.
//   dword index = ((ntile*KTILES + kt)*32 + lane)*8 + v
//   lane -> N = ntile*16 + lane%16 ; v -> K pair (k, k+1), k = kt*32 + 8*(lane/16) + 2v + 8*(v/4)
// ---------------------------------------------------------------------------
__global__ __launch_bounds__(256)
void pack_w_bf16(const float* __restrict__ W, uint32_t* __restrict__ out) {
  int gid   = blockIdx.x * blockDim.x + threadIdx.x;
  int v     = gid & 7;
  int lane  = (gid >> 3) & 31;
  int kt    = (gid >> 8) & 31;
  int ntile = gid >> 13;
  int k  = kt * 32 + 8 * (lane >> 4) + 2 * v + ((v >> 2) << 3);
  int n0 = ntile * 16 + (lane & 15);
  union { __bf16 h[2]; uint32_t u; } p;
  p.h[0] = (__bf16)W[(size_t)n0 * N_ + k];
  p.h[1] = (__bf16)W[(size_t)n0 * N_ + k + 1];
  out[gid] = p.u;
}

__global__ void barrier_init(uint32_t* bar) { bar[0] = 0; bar[1] = 0; }

// device-scope grid barrier: arrive counter + generation, s_sleep spin
__device__ __forceinline__ void grid_barrier(uint32_t* cnt, uint32_t* gen) {
  __syncthreads();
  if (threadIdx.x == 0) {
    __threadfence();   // make this block's y stores device-visible
    uint32_t g = __hip_atomic_load(gen, __ATOMIC_RELAXED, __HIP_MEMORY_SCOPE_AGENT);
    uint32_t a = __hip_atomic_fetch_add(cnt, 1u, __ATOMIC_ACQ_REL, __HIP_MEMORY_SCOPE_AGENT);
    if (a == NBLOCKS - 1) {
      __hip_atomic_store(cnt, 0u, __ATOMIC_RELAXED, __HIP_MEMORY_SCOPE_AGENT);
      __hip_atomic_fetch_add(gen, 1u, __ATOMIC_ACQ_REL, __HIP_MEMORY_SCOPE_AGENT);
    } else {
      while (__hip_atomic_load(gen, __ATOMIC_ACQUIRE, __HIP_MEMORY_SCOPE_AGENT) == g)
        __builtin_amdgcn_s_sleep(2);
    }
  }
  __syncthreads();
}

// ---------------------------------------------------------------------------
// Persistent recurrence kernel. Grid (16,8) x 128 threads (4 waves).
//   blockIdx.y -> batch tile (16 rows); wave -> one 16-wide n-tile.
// ---------------------------------------------------------------------------
__global__ __launch_bounds__(128)
void rnn_persistent(const float* __restrict__ y0,
                    float* __restrict__ yseq,
                    const float* __restrict__ noise,
                    const float* __restrict__ u,
                    const float* __restrict__ W_in_raw,
                    const float* __restrict__ b_rec,
                    const float* __restrict__ W_gate_in,
                    const float* __restrict__ b_gate,
                    const uint32_t* __restrict__ packR,
                    const uint32_t* __restrict__ packG,
                    uint32_t* __restrict__ bar)
{
  extern __shared__ char smem[];
  __bf16 (*sY)[N_ + 8] = (__bf16(*)[N_ + 8])smem;
  float  (*sU)[8]      = (float(*)[8])(smem + SU_OFF);

  const int tid   = threadIdx.x;
  const int lane  = tid & 31;
  const int wave  = tid >> 5;
  const int mtile = blockIdx.y;
  const int ntile = blockIdx.x * 4 + wave;
  const int b0    = mtile * 16;
  const int hi    = lane >> 4;      // lane-half -> K sub-range of A fragment
  const int am    = lane & 15;      // A-matrix row (M)

  // per-wave LDS weight strip: [R half0 | R half1 | G half0 | G half1], 16KB each
  uint4* lR0 = (uint4*)(smem + SW_OFF + wave * SW_PER_WAVE);
  uint4* lR1 = lR0 + 1024;
  uint4* lG0 = lR1 + 1024;
  uint4* lG1 = lG0 + 1024;

  // ---- one-time: stage packed weight strips L2 -> LDS (resident all T) ----
  {
    const uint4* gR = (const uint4*)(packR + ((size_t)ntile * KTILES * 32) * 8);
    const uint4* gG = (const uint4*)(packG + ((size_t)ntile * KTILES * 32) * 8);
    for (int kt = 0; kt < KTILES; ++kt) {
      int idx = kt * 32 + lane;
      lR0[idx] = gR[idx * 2];
      lR1[idx] = gR[idx * 2 + 1];
      lG0[idx] = gG[idx * 2];
      lG1[idx] = gG[idx * 2 + 1];
    }
  }

  // ---- hoist t-invariant epilogue constants (per lane: one n column) ----
  const int n    = ntile * 16 + (lane & 15);
  const float brn = b_rec[n];
  const float bgn = b_gate[n];
  float win[NI_], wgin[NI_];
#pragma unroll
  for (int i = 0; i < NI_; ++i) {
    win[i]  = fabsf(W_in_raw[(size_t)n * NI_ + i]);   // W_in = |W_in_raw|
    wgin[i] = W_gate_in[(size_t)n * NI_ + i];
  }

  uint32_t* cnt = bar;
  uint32_t* gen = bar + 1;

  union Frag { v16bf v; v8bf h[2]; };

  for (int t = 0; t < T_; ++t) {
    const float* yp = (t == 0) ? y0 : (yseq + (size_t)(t - 1) * N_);
    const long   ps = (t == 0) ? (long)N_ : (long)T_ * N_;

    // ---- load y_prev tile (16 x 1024 fp32) -> bf16 LDS ----
    for (int kk = 0; kk < 32; ++kk) {
      int j    = kk * 128 + tid;      // float4 index in [0,4096)
      int row  = j >> 8;
      int col4 = j & 255;
      const float4 y4 = *(const float4*)(yp + (size_t)(b0 + row) * ps + col4 * 4);
      union { __bf16 h[4]; uint2 u2; } pk;
      pk.h[0] = (__bf16)y4.x; pk.h[1] = (__bf16)y4.y;
      pk.h[2] = (__bf16)y4.z; pk.h[3] = (__bf16)y4.w;
      *(uint2*)&sY[row][col4 * 4] = pk.u2;
    }
    if (tid < 16 * NI_) {
      int row = tid / NI_, i = tid % NI_;
      sU[row][i] = u[((size_t)(b0 + row) * T_ + t) * NI_ + i];
    }
    __syncthreads();

    // ---- dual GEMM over K, double-buffered: issue kt+1 loads before
    //      kt's WMMAs so ds_load latency overlaps matrix math ----
    v8f accR = {};
    v8f accG = {};
    Frag A[2], Br[2], Bg[2];

    // prime kt = 0
    A[0].h[0]  = *(const v8bf*)&sY[am][hi * 8];
    A[0].h[1]  = *(const v8bf*)&sY[am][16 + hi * 8];
    Br[0].h[0] = ((const v8bf*)lR0)[lane];
    Br[0].h[1] = ((const v8bf*)lR1)[lane];
    Bg[0].h[0] = ((const v8bf*)lG0)[lane];
    Bg[0].h[1] = ((const v8bf*)lG1)[lane];

#pragma unroll
    for (int kt = 0; kt < KTILES; ++kt) {
      const int cur = kt & 1, nxt = cur ^ 1;
      if (kt + 1 < KTILES) {
        const int idx = (kt + 1) * 32 + lane;
        A[nxt].h[0]  = *(const v8bf*)&sY[am][(kt + 1) * 32 + hi * 8];
        A[nxt].h[1]  = *(const v8bf*)&sY[am][(kt + 1) * 32 + 16 + hi * 8];
        Br[nxt].h[0] = ((const v8bf*)lR0)[idx];
        Br[nxt].h[1] = ((const v8bf*)lR1)[idx];
        Bg[nxt].h[0] = ((const v8bf*)lG0)[idx];
        Bg[nxt].h[1] = ((const v8bf*)lG1)[idx];
      }
      accR = __builtin_amdgcn_wmma_f32_16x16x32_bf16(false, A[cur].v, false, Br[cur].v,
                                                     (short)0, accR, false, false);
      accG = __builtin_amdgcn_wmma_f32_16x16x32_bf16(false, A[cur].v, false, Bg[cur].v,
                                                     (short)0, accG, false, false);
    }

    // ---- epilogue: C/D layout VGPR r -> M (= r + 8*hi), lane&15 -> N ----
#pragma unroll
    for (int r = 0; r < 8; ++r) {
      int ml = r + hi * 8;
      int b  = b0 + ml;
      float wu = brn, wug = bgn;
#pragma unroll
      for (int i = 0; i < NI_; ++i) {
        wu  += sU[ml][i] * win[i];
        wug += sU[ml][i] * wgin[i];
      }
      float gate = 1.0f / (1.0f + __expf(-(accG[r] + wug)));
      float pre  = accR[r] + wu;
      float rel  = pre > 0.0f ? pre : 0.0f;
      float yold = yp[(size_t)b * ps + n];
      float nz   = noise[((size_t)b * T_ + t) * N_ + n];
      float ynew = gate * yold + (1.0f - gate) * rel + NOISE_SCALE * nz;
      yseq[((size_t)b * T_ + t) * N_ + n] = ynew;
    }

    grid_barrier(cnt, gen);   // publish y_t, wait for all blocks
  }
}

// ---------------------------------------------------------------------------
// Readout: z[b,t] = sigmoid(y_seq[b,t,:] . w_out + b_out). One wave per (b,t).
// ---------------------------------------------------------------------------
__global__ __launch_bounds__(256)
void out_kernel(const float* __restrict__ yseq, const float* __restrict__ w_out,
                const float* __restrict__ b_out, float* __restrict__ z,
                float* __restrict__ yfinal)
{
  int wave = threadIdx.x >> 5, lane = threadIdx.x & 31;
  int p = blockIdx.x * 8 + wave;           // (b,t) pair, p < B*T
  int b = p / T_, t = p % T_;
  const float* y = yseq + ((size_t)b * T_ + t) * N_;
  float s = 0.0f;
  for (int k = lane; k < N_; k += 32) s += y[k] * w_out[k];
#pragma unroll
  for (int off = 16; off > 0; off >>= 1) s += __shfl_down(s, off, 32);
  if (lane == 0)
    z[(size_t)b * T_ + t] = 1.0f / (1.0f + __expf(-(s + b_out[0])));
  if (t == T_ - 1) {
    for (int k = lane; k < N_; k += 32) yfinal[(size_t)b * N_ + k] = y[k];
  }
}

// ---------------------------------------------------------------------------
extern "C" void kernel_launch(void* const* d_in, const int* in_sizes, int n_in,
                              void* d_out, int out_size, void* d_ws, size_t ws_size,
                              hipStream_t stream) {
  (void)in_sizes; (void)n_in; (void)out_size; (void)ws_size;

  const float* y0      = (const float*)d_in[0];
  const float* u       = (const float*)d_in[1];
  const float* noise   = (const float*)d_in[2];
  const float* Winraw  = (const float*)d_in[3];
  const float* Wrec    = (const float*)d_in[4];
  const float* brec    = (const float*)d_in[5];
  const float* wout    = (const float*)d_in[6];
  const float* bout    = (const float*)d_in[7];
  const float* Wgate   = (const float*)d_in[8];
  const float* Wgatein = (const float*)d_in[9];
  const float* bgate   = (const float*)d_in[10];

  float* yseq = (float*)d_out;                       // [B,T,N]
  float* z    = yseq + (size_t)B_ * T_ * N_;         // [B,T,1]
  float* yfin = z + (size_t)B_ * T_;                 // [B,N]

  const size_t packDwords = (size_t)NTILES * KTILES * 32 * 8;  // 512K dwords
  uint32_t* packR = (uint32_t*)d_ws;                 // 2 MB
  uint32_t* packG = packR + packDwords;              // 2 MB
  uint32_t* bar   = packG + packDwords;              // barrier cnt/gen

  // allow >64KB dynamic LDS for the persistent kernel (not a stream op)
  static_assert(SMEM_BYTES <= 320 * 1024, "exceeds CDNA5 LDS per workgroup");
  (void)hipFuncSetAttribute((const void*)rnn_persistent,
                            hipFuncAttributeMaxDynamicSharedMemorySize,
                            (int)SMEM_BYTES);

  // one-time: pack weights into bf16 WMMA B-fragment layout
  pack_w_bf16<<<(int)(packDwords / 256), 256, 0, stream>>>(Wrec, packR);
  pack_w_bf16<<<(int)(packDwords / 256), 256, 0, stream>>>(Wgate, packG);
  barrier_init<<<1, 1, 0, stream>>>(bar);

  // whole 512-step recurrence in ONE launch; grid barrier between steps
  dim3 grid(GRID_X, GRID_Y);
  rnn_persistent<<<grid, 128, SMEM_BYTES, stream>>>(y0, yseq, noise, u,
                                                    Winraw, brec, Wgatein, bgate,
                                                    packR, packG, bar);

  // readout pass (z never feeds the recurrence)
  out_kernel<<<(B_ * T_) / 8, 256, 0, stream>>>(yseq, wout, bout, z, yfin);
}
